// glimpse_62182536511585
// MI455X (gfx1250) — matrix operand order repaired and checked by reading, not compiled, not run on hardware
//
#include <hip/hip_runtime.h>

typedef __attribute__((ext_vector_type(16))) _Float16 v16h;
typedef __attribute__((ext_vector_type(8)))  float    v8f;

#define N_DET 182
#define N_ANG 180
#define FPAD  512
#define WSZ   9
#define ROWS  2048            // B * B_PIXELS
#define KACT  14580           // N_ANG * 81
#define KPAD0 14592           // next multiple of 32

// ---------------------------------------------------------------------------
// 1) Precompute g = Re(ifft(fourier_filter)) and cos/sin of projection angles.
__global__ void prep_kernel(const float* __restrict__ F, const float* __restrict__ theta,
                            float* __restrict__ gr, float* __restrict__ cs,
                            float* __restrict__ sn) {
  int m = blockIdx.x * blockDim.x + threadIdx.x;
  if (m < FPAD) {
    double acc = 0.0;
    for (int k = 0; k < FPAD; ++k) {
      int ph = (k * m) & (FPAD - 1);                     // exact phase reduction
      acc += (double)F[k] * cos(6.2831853071795864769 * (double)ph / (double)FPAD);
    }
    gr[m] = (float)(acc / (double)FPAD);
  }
  if (m < N_ANG) {
    float th = theta[m];
    cs[m] = cosf(th);
    sn[m] = sinf(th);
  }
}

// ---------------------------------------------------------------------------
// 2) Ramp filtering as circular convolution; writes col[b][a][det] (transposed).
__global__ __launch_bounds__(256) void filter_kernel(const float* __restrict__ sino,
                                                     const float* __restrict__ gr,
                                                     float* __restrict__ col) {
  __shared__ float ssin[184];
  __shared__ float sg[FPAD];
  int b = blockIdx.x / N_ANG, a = blockIdx.x % N_ANG;
  int t = threadIdx.x;
  for (int j = t; j < N_DET; j += 256) ssin[j] = sino[((size_t)b * N_DET + j) * N_ANG + a];
  for (int j = t; j < FPAD;  j += 256) sg[j] = gr[j];
  __syncthreads();
  if (t < N_DET) {
    float s = 0.f;
    for (int j = 0; j < N_DET; ++j) s += ssin[j] * sg[(t - j) & (FPAD - 1)];
    col[((size_t)b * N_ANG + a) * N_DET + t] = s;
  }
}

// ---------------------------------------------------------------------------
// 3) Patch backprojection (grid sample degenerates to 1-D lerp along detector).
//    Builds f16 activation matrix X[2048][14592], zero-padded K tail.
__global__ __launch_bounds__(256) void bp_kernel(const float* __restrict__ coord,
                                                 const float* __restrict__ col,
                                                 const float* __restrict__ cs,
                                                 const float* __restrict__ sn,
                                                 const float* __restrict__ sS,
                                                 const float* __restrict__ w1,
                                                 const float* __restrict__ w2,
                                                 _Float16* __restrict__ X) {
  int idx = blockIdx.x * blockDim.x + threadIdx.x;     // < 2048*14592 < 2^31
  int row = idx / KPAD0;
  int k   = idx - row * KPAD0;
  if (k >= KACT) { X[idx] = (_Float16)0.f; return; }
  int a = k / 81, r = k - a * 81, wy = r / 9, wx = r - wy * 9;
  float cy = coord[row * 2 + 0];
  float cx = coord[row * 2 + 1];
  const float cr2 = 8.0f / 127.0f;                     // crop/2 (h = 128)
  float gx = cr2 * w1[0] * (-1.0f + 0.25f * (float)wx) + 2.0f * cx;
  float gy = cr2 * w2[0] * (-1.0f + 0.25f * (float)wy) + 2.0f * cy;
  float c0 = (gy * 0.5f + 0.5f) * 127.0f;
  float c1 = (gx * 0.5f + 0.5f) * 127.0f;
  c0 = (c0 > 127.5f) ? 255.0f - c0 : c0;  c0 = (c0 < -0.5f) ? -1.0f - c0 : c0;
  c1 = (c1 > 127.5f) ? 255.0f - c1 : c1;  c1 = (c1 < -0.5f) ? -1.0f - c1 : c1;
  float s   = sS[0];
  float xpr = (c0 - 63.5f) / s;
  float ypr = (c1 - 63.5f) / s;
  float tt  = ypr * cs[a] - xpr * sn[a];
  float ix  = (tt + 1.0f) * 0.5f * 181.0f;
  float x0f = floorf(ix);
  float wf  = ix - x0f;
  int   x0  = (int)x0f;
  int   b   = row >> 10;
  const float* crow = col + ((size_t)b * N_ANG + a) * N_DET;
  float v0 = (x0 >= 0     && x0     <= 181) ? crow[x0]     : 0.f;
  float v1 = (x0 + 1 >= 0 && x0 + 1 <= 181) ? crow[x0 + 1] : 0.f;
  X[idx] = (_Float16)(v0 * (1.f - wf) + v1 * wf);
}

// ---------------------------------------------------------------------------
// 4) Pack fp32 weights (K x N, row-major) into f16 WMMA-B block layout:
//    out[((k2>>4)*2 + ((k2>>3)&1))*(N*8) + n*8 + (k2&7)] = {W[2k2][n], W[2k2+1][n]}
__global__ void pack_kernel(const float* __restrict__ Wm, unsigned* __restrict__ out,
                            int K, int Kpad, int N) {
  int idx = blockIdx.x * blockDim.x + threadIdx.x;
  int tot = (Kpad >> 1) * N;
  if (idx >= tot) return;
  int k2 = idx / N, n = idx - k2 * N;
  float w0 = (2 * k2     < K) ? Wm[(size_t)(2 * k2)     * N + n] : 0.f;
  float w1 = (2 * k2 + 1 < K) ? Wm[(size_t)(2 * k2 + 1) * N + n] : 0.f;
  union { _Float16 h[2]; unsigned u; } pk;
  pk.h[0] = (_Float16)w0;
  pk.h[1] = (_Float16)w1;
  unsigned dst = (unsigned)(((k2 >> 4) * 2 + ((k2 >> 3) & 1)) * (N * 8) + n * 8 + (k2 & 7));
  out[dst] = pk.u;
}

// ---------------------------------------------------------------------------
// 5) WMMA GEMM: C = relu(A(MxK,f16,row-major,lda) * B(packed) + bias), C f16.
//    Each wave computes a 16x32 output strip (two 16x16 tiles): the A fragment
//    is loaded once per K-step and reused for two WMMAs (3 b128 loads / wmma).
__global__ __launch_bounds__(128) void gemm_wmma(const _Float16* __restrict__ A,
                                                 const unsigned* __restrict__ Bp,
                                                 const float* __restrict__ bias,
                                                 _Float16* __restrict__ C,
                                                 int M, int N, int K, int lda) {
  int wave = blockIdx.x * (blockDim.x >> 5) + (threadIdx.x >> 5);
  int lane = threadIdx.x & 31;
  int nt = N >> 5;                 // number of 32-wide column strips
  int tm = wave / nt, tn = wave - tm * nt;
  if (tm >= (M >> 4)) return;
  int hl  = lane >> 4;             // lane half (0: lanes 0-15, 1: lanes 16-31)
  int l15 = lane & 15;
  int n0  = (tn << 5) + l15;       // column of first tile for this lane
  int n1  = n0 + 16;               // column of second tile
  const _Float16* Arow = A + (size_t)((tm << 4) + l15) * lda;
  union Frag { v16h v; uint4 q[2]; };
  v8f acc0 = {};
  v8f acc1 = {};
  for (int kb = 0; kb < K; kb += 32) {
    Frag fa, fb0, fb1;
    // A fragment: lanes 0-15 hold K = kb+0..7 (q0) and kb+8..15 (q1); +16 for hl=1
    fa.q[0] = *(const uint4*)(Arow + kb + 16 * hl);
    fa.q[1] = *(const uint4*)(Arow + kb + 16 * hl + 8);
    // B fragments: packed contiguous 8 dwords per (kblock, half, column)
    size_t bbase = (size_t)((kb >> 5) * 2 + hl) * N;
    const uint4* bq0 = (const uint4*)(Bp + (bbase + n0) * 8);
    const uint4* bq1 = (const uint4*)(Bp + (bbase + n1) * 8);
    fb0.q[0] = bq0[0];
    fb0.q[1] = bq0[1];
    fb1.q[0] = bq1[0];
    fb1.q[1] = bq1[1];
    acc0 = __builtin_amdgcn_wmma_f32_16x16x32_f16(false, fa.v, false, fb0.v,
                                                  (short)0, acc0, false, false);
    acc1 = __builtin_amdgcn_wmma_f32_16x16x32_f16(false, fa.v, false, fb1.v,
                                                  (short)0, acc1, false, false);
  }
  float bn0 = bias[n0];
  float bn1 = bias[n1];
#pragma unroll
  for (int r = 0; r < 8; ++r) {
    int m = (tm << 4) + r + 8 * hl;       // C/D layout: VGPR r -> M = r + 8*hl
    float v0 = acc0[r] + bn0;
    float v1 = acc1[r] + bn1;
    v0 = v0 > 0.f ? v0 : 0.f;
    v1 = v1 > 0.f ? v1 : 0.f;
    C[(size_t)m * N + n0] = (_Float16)v0;
    C[(size_t)m * N + n1] = (_Float16)v1;
  }
}

// ---------------------------------------------------------------------------
// 6) Final 64 -> 1 linear (no relu) scaled by pi/2, fp32 output.
__global__ void final_kernel(const _Float16* __restrict__ act, const float* __restrict__ Wm,
                             const float* __restrict__ bias, float* __restrict__ out, int M) {
  int row = blockIdx.x * blockDim.x + threadIdx.x;
  if (row >= M) return;
  float s = 0.f;
  for (int k = 0; k < 64; ++k) s += (float)act[row * 64 + k] * Wm[k];
  out[row] = (s + bias[0]) * 1.5707963267948966f;
}

// ---------------------------------------------------------------------------
extern "C" void kernel_launch(void* const* d_in, const int* in_sizes, int n_in,
                              void* d_out, int out_size, void* d_ws, size_t ws_size,
                              hipStream_t stream) {
  (void)in_sizes; (void)n_in; (void)out_size; (void)ws_size;
  const float* coord = (const float*)d_in[0];
  const float* sino  = (const float*)d_in[1];
  const float* ff    = (const float*)d_in[2];
  const float* sS    = (const float*)d_in[3];
  // d_in[4] = z (unused: iy == angle index exactly)
  const float* theta = (const float*)d_in[5];
  const float* w1s   = (const float*)d_in[6];
  const float* w2s   = (const float*)d_in[7];
  const float* Wm[10]; const float* Bb[10];
  for (int i = 0; i < 10; ++i) { Wm[i] = (const float*)d_in[8 + i]; Bb[i] = (const float*)d_in[18 + i]; }

  static const int Kd[10] = {KACT, 256, 256, 256, 256, 128, 128, 128, 64, 64};
  static const int Nd[10] = {256, 256, 256, 256, 128, 128, 128, 64, 64, 1};
  static const int Kp[9]  = {KPAD0, 256, 256, 256, 256, 128, 128, 128, 64};

  char* base = (char*)d_ws;
  size_t cur = 0;
  auto carve = [&](size_t bytes) -> void* {
    void* p = base + cur;
    cur = (cur + bytes + 255) & ~(size_t)255;
    return p;
  };
  float*    gr  = (float*)carve(FPAD * sizeof(float));
  float*    cs  = (float*)carve(N_ANG * sizeof(float));
  float*    sn  = (float*)carve(N_ANG * sizeof(float));
  float*    col = (float*)carve((size_t)2 * N_ANG * N_DET * sizeof(float));
  _Float16* X   = (_Float16*)carve((size_t)ROWS * KPAD0 * sizeof(_Float16));
  unsigned* Bp[9];
  for (int l = 0; l < 9; ++l) Bp[l] = (unsigned*)carve((size_t)(Kp[l] / 2) * Nd[l] * sizeof(unsigned));
  _Float16* act[9];
  for (int l = 0; l < 9; ++l) act[l] = (_Float16*)carve((size_t)ROWS * Nd[l] * sizeof(_Float16));

  prep_kernel<<<2, 256, 0, stream>>>(ff, theta, gr, cs, sn);
  filter_kernel<<<2 * N_ANG, 256, 0, stream>>>(sino, gr, col);
  bp_kernel<<<(ROWS * KPAD0) / 256, 256, 0, stream>>>(coord, col, cs, sn, sS, w1s, w2s, X);
  for (int l = 0; l < 9; ++l) {
    int tot = (Kp[l] / 2) * Nd[l];
    pack_kernel<<<(tot + 255) / 256, 256, 0, stream>>>(Wm[l], Bp[l], Kd[l], Kp[l], Nd[l]);
  }
  for (int l = 0; l < 9; ++l) {
    const _Float16* Aact = (l == 0) ? (const _Float16*)X : (const _Float16*)act[l - 1];
    int lda   = (l == 0) ? KPAD0 : Kd[l];
    int waves = (ROWS / 16) * (Nd[l] / 32);   // one wave per 16x32 strip
    gemm_wmma<<<waves / 4, 128, 0, stream>>>(Aact, Bp[l], Bb[l], act[l], ROWS, Nd[l], Kp[l], lda);
  }
  final_kernel<<<ROWS / 256, 256, 0, stream>>>(act[8], Wm[9], Bb[9], (float*)d_out, ROWS);
}
